// Multi_Head_Attention_20925080666716
// MI455X (gfx1250) — matrix operand order repaired and checked by reading, not compiled, not run on hardware
//
#include <hip/hip_runtime.h>

// ---------------- problem constants ----------------
#define BB   8
#define SS   1024
#define DD   1024
#define HH   16
#define DHD  64
#define MTOT (BB*SS)          // 8192 rows
#define GRP  (BB*HH)          // 128 "heads" after raw reshape
#define GELEM 65536           // elems per head block (1024*64)
#define LN_EPS 1e-5f

typedef _Float16 v16h __attribute__((ext_vector_type(16)));
typedef float    v8f  __attribute__((ext_vector_type(8)));
typedef unsigned int v4u  __attribute__((ext_vector_type(4)));
typedef int          v8i_ __attribute__((ext_vector_type(8)));
typedef int          v4i_ __attribute__((ext_vector_type(4)));

union F16x8 { v16h v; float4 f[2]; };
union PackH { _Float16 h[8]; float4 f; };

// ---------------- TDM: async 2-D f16 tile load into LDS ----------------
// Builds a Tensor DMA Descriptor (D#) per CDNA5 ISA §8:
//  group0: [1:0]=count=1, [63:32]=lds_addr, [120:64]=global_addr, [127:126]=type=2
//  group1: data_size=1 (2B), tensor_dim0/1, tile_dim0/1, tensor_dim0_stride
__device__ __forceinline__ unsigned lds_addr_of(const void* p) {
  // generic pointer into LDS aperture: low 32 bits are the LDS byte address
  return (unsigned)(uintptr_t)p;
}

__device__ __forceinline__ void tdm_load_tile_f16(
    unsigned lds_byte_addr, const _Float16* gptr,
    unsigned tensor_w, unsigned tensor_h, unsigned row_stride_elems,
    unsigned tile_w, unsigned tile_h) {
  unsigned long long ga = (unsigned long long)(uintptr_t)gptr;
  v4u g0;
  g0[0] = 1u;                                              // count=1 (valid user D#)
  g0[1] = lds_byte_addr;                                   // lds_addr
  g0[2] = (unsigned)(ga & 0xFFFFFFFFu);                    // global_addr[31:0]
  g0[3] = (unsigned)((ga >> 32) & 0x01FFFFFFu) | (2u << 30); // addr[56:32] | type=2
  v8i_ g1;
  g1[0] = (int)(1u << 16);                                 // data_size=1 -> 2 bytes
  g1[1] = (int)((tensor_w & 0xFFFFu) << 16);               // tensor_dim0[15:0]
  g1[2] = (int)((tensor_w >> 16) | ((tensor_h & 0xFFFFu) << 16)); // dim0 hi | dim1 lo
  g1[3] = (int)((tensor_h >> 16) | (tile_w << 16));        // dim1 hi | tile_dim0
  g1[4] = (int)(tile_h & 0xFFFFu);                         // tile_dim1 (tile_dim2=0)
  g1[5] = (int)row_stride_elems;                           // tensor_dim0_stride[31:0]
  g1[6] = 0;                                               // stride[47:32], dim1_stride lo
  g1[7] = 0;
  v4i_ z4 = {0, 0, 0, 0};                                  // groups 2/3 unused (2-D)
  v8i_ z8 = {0, 0, 0, 0, 0, 0, 0, 0};
  __builtin_amdgcn_tensor_load_to_lds(g0, g1, z4, z4, z8, 0);
}

// ---------------- fp32 -> f16 convert ----------------
__global__ __launch_bounds__(256) void cvt_f32_f16(const float* __restrict__ s,
                                                   _Float16* __restrict__ d, int n) {
  int i = blockIdx.x * blockDim.x + threadIdx.x;
  int stride = gridDim.x * blockDim.x;
  for (; i < n; i += stride) d[i] = (_Float16)s[i];
}

// ---------------- fused QKV projection (TDM double-buffered) ----------------
// Y = Xh @ Wh[z]^T + b[z] ; z=0 -> Qh, z=1 -> Kh, z=2 -> Vt (transposed-in-group)
__global__ __launch_bounds__(256) void qkv_gemm(
    const _Float16* __restrict__ X,   // [8192,1024] f16
    const _Float16* __restrict__ Wh,  // [3][1024][1024] f16 (q,k,v contiguous)
    const float* __restrict__ bq, const float* __restrict__ bk, const float* __restrict__ bv,
    _Float16* __restrict__ Qh, _Float16* __restrict__ Kh, _Float16* __restrict__ Vt) {
  const int z = blockIdx.z;
  const _Float16* W = Wh + (size_t)z * DD * DD;
  const float* bias = (z == 0) ? bq : (z == 1) ? bk : bv;
  const int tm = blockIdx.x * 128;
  const int tn = blockIdx.y * 128;

  __shared__ alignas(16) _Float16 As[2][128 * 32];
  __shared__ alignas(16) _Float16 Bs[2][128 * 32];

  const int t = threadIdx.x;
  const int lane = t & 31, wave = t >> 5;
  const int wm = (wave >> 1) * 32;   // 0,32,64,96
  const int wn = (wave & 1) * 64;    // 0,64
  const int hi = (lane >= 16) ? 1 : 0;
  const int l16 = lane & 15;

  v8f acc[2][4];
  for (int i = 0; i < 2; ++i) for (int j = 0; j < 4; ++j) acc[i][j] = (v8f)0.0f;

  if (wave == 0) {  // prologue: DMA tile 0 into buffer 0
    tdm_load_tile_f16(lds_addr_of(As[0]), &X[(size_t)tm * DD], DD, MTOT, DD, 32, 128);
    tdm_load_tile_f16(lds_addr_of(Bs[0]), &W[(size_t)tn * DD], DD, DD,   DD, 32, 128);
  }

  for (int it = 0; it < DD / 32; ++it) {
    const int k0 = it * 32;
    const int cur = it & 1;
    if (wave == 0) {
      if (k0 + 32 < DD) {  // DMA next tile into other buffer, wait current pair
        tdm_load_tile_f16(lds_addr_of(As[cur ^ 1]), &X[(size_t)tm * DD + k0 + 32], DD, MTOT, DD, 32, 128);
        tdm_load_tile_f16(lds_addr_of(Bs[cur ^ 1]), &W[(size_t)tn * DD + k0 + 32], DD, DD,   DD, 32, 128);
        __builtin_amdgcn_s_wait_tensorcnt(2);
      } else {
        __builtin_amdgcn_s_wait_tensorcnt(0);
      }
    }
    __syncthreads();

    v16h afrag[2], bfrag[4];
    for (int i = 0; i < 2; ++i) {
      const _Float16* p = &As[cur][(wm + i * 16 + l16) * 32];
      F16x8 u; u.f[0] = *(const float4*)(p + hi * 8);
               u.f[1] = *(const float4*)(p + 16 + hi * 8);
      afrag[i] = u.v;
    }
    for (int j = 0; j < 4; ++j) {
      const _Float16* p = &Bs[cur][(wn + j * 16 + l16) * 32];
      F16x8 u; u.f[0] = *(const float4*)(p + hi * 8);
               u.f[1] = *(const float4*)(p + 16 + hi * 8);
      bfrag[j] = u.v;
    }
    for (int i = 0; i < 2; ++i)
      for (int j = 0; j < 4; ++j)
        acc[i][j] = __builtin_amdgcn_wmma_f32_16x16x32_f16(
            false, afrag[i], false, bfrag[j], (short)0, acc[i][j], false, false);
    __syncthreads();   // readers done before next TDM overwrites this buffer
  }

  for (int i = 0; i < 2; ++i)
    for (int j = 0; j < 4; ++j) {
      int col = tn + wn + j * 16 + l16;
      float bcol = bias[col];
      for (int r = 0; r < 8; ++r) {
        int row = tm + wm + i * 16 + r + hi * 8;
        float v = acc[i][j][r] + bcol;
        if (z == 0) {
          Qh[(size_t)row * DD + col] = (_Float16)v;
        } else if (z == 1) {
          Kh[(size_t)row * DD + col] = (_Float16)v;
        } else {
          // raw-reshape head block: g = row/64, r' = (row%64)*16 + col/64, c' = col%64
          int g  = row >> 6;
          int rr = ((row & 63) << 4) + (col >> 6);
          int cc = col & 63;
          Vt[(size_t)g * GELEM + (size_t)cc * SS + rr] = (_Float16)v;  // Vt[g][d][j]
        }
      }
    }
}

// ---------------- scores = Q K^T * scale ----------------
__global__ __launch_bounds__(256) void scores_kernel(
    const _Float16* __restrict__ Qh, const _Float16* __restrict__ Kh,
    float* __restrict__ alpha) {
  const int g = blockIdx.z;
  const int tq = blockIdx.x * 128;
  const int tk = blockIdx.y * 128;
  const _Float16* Q = Qh + (size_t)g * GELEM;  // [1024][64]
  const _Float16* K = Kh + (size_t)g * GELEM;  // [1024][64]

  __shared__ alignas(16) _Float16 As[128 * 32];
  __shared__ alignas(16) _Float16 Bs[128 * 32];

  const int t = threadIdx.x;
  const int lane = t & 31, wave = t >> 5;
  const int wm = (wave >> 1) * 32;
  const int wn = (wave & 1) * 64;
  const int hi = (lane >= 16) ? 1 : 0;
  const int l16 = lane & 15;

  v8f acc[2][4];
  for (int i = 0; i < 2; ++i) for (int j = 0; j < 4; ++j) acc[i][j] = (v8f)0.0f;

  for (int k0 = 0; k0 < DHD; k0 += 32) {
    for (int i = 0; i < 2; ++i) {
      int c = t + i * 256;
      int row = c >> 2, off = (c & 3) << 3;
      *(float4*)&As[row * 32 + off] = *(const float4*)&Q[(size_t)(tq + row) * DHD + k0 + off];
      *(float4*)&Bs[row * 32 + off] = *(const float4*)&K[(size_t)(tk + row) * DHD + k0 + off];
    }
    __syncthreads();

    v16h afrag[2], bfrag[4];
    for (int i = 0; i < 2; ++i) {
      const _Float16* p = &As[(wm + i * 16 + l16) * 32];
      F16x8 u; u.f[0] = *(const float4*)(p + hi * 8);
               u.f[1] = *(const float4*)(p + 16 + hi * 8);
      afrag[i] = u.v;
    }
    for (int j = 0; j < 4; ++j) {
      const _Float16* p = &Bs[(wn + j * 16 + l16) * 32];
      F16x8 u; u.f[0] = *(const float4*)(p + hi * 8);
               u.f[1] = *(const float4*)(p + 16 + hi * 8);
      bfrag[j] = u.v;
    }
    for (int i = 0; i < 2; ++i)
      for (int j = 0; j < 4; ++j)
        acc[i][j] = __builtin_amdgcn_wmma_f32_16x16x32_f16(
            false, afrag[i], false, bfrag[j], (short)0, acc[i][j], false, false);
    __syncthreads();
  }

  float* out = alpha + (size_t)g * SS * SS;
  for (int i = 0; i < 2; ++i)
    for (int j = 0; j < 4; ++j) {
      int col = tk + wn + j * 16 + l16;
      for (int r = 0; r < 8; ++r) {
        int row = tq + wm + i * 16 + r + hi * 8;
        out[(size_t)row * SS + col] = acc[i][j][r] * 0.125f;  // DH^-0.5
      }
    }
}

// ---------------- row softmax in-place on alpha ----------------
__global__ __launch_bounds__(256) void softmax_kernel(float* __restrict__ alpha) {
  int row = blockIdx.x * 8 + (threadIdx.x >> 5);  // one wave per row
  int lane = threadIdx.x & 31;
  float* p = alpha + (size_t)row * SS;
  float v[32];
  float mx = -3.0e38f;
  for (int i = 0; i < 32; ++i) { v[i] = p[lane + i * 32]; mx = fmaxf(mx, v[i]); }
  for (int o = 16; o; o >>= 1) mx = fmaxf(mx, __shfl_xor(mx, o, 32));
  float sum = 0.f;
  for (int i = 0; i < 32; ++i) { v[i] = __expf(v[i] - mx); sum += v[i]; }
  for (int o = 16; o; o >>= 1) sum += __shfl_xor(sum, o, 32);
  float inv = 1.0f / sum;
  for (int i = 0; i < 32; ++i) p[lane + i * 32] = v[i] * inv;
}

// ---------------- context = alpha @ V ----------------
__global__ __launch_bounds__(256) void pv_kernel(
    const float* __restrict__ alpha, const _Float16* __restrict__ Vt,
    _Float16* __restrict__ Ch) {
  const int g = blockIdx.y;
  const int tq = blockIdx.x * 128;
  const float*    P = alpha + (size_t)g * SS * SS;  // [1024][1024] fp32
  const _Float16* V = Vt    + (size_t)g * GELEM;    // [64][1024]  (d-major)

  __shared__ alignas(16) _Float16 As[128 * 32];
  __shared__ alignas(16) _Float16 Bs[64 * 32];

  const int t = threadIdx.x;
  const int lane = t & 31, wave = t >> 5;
  const int wq = wave * 16;   // 8 waves x 16 rows = 128 rows
  const int hi = (lane >= 16) ? 1 : 0;
  const int l16 = lane & 15;

  v8f acc[4];
  for (int j = 0; j < 4; ++j) acc[j] = (v8f)0.0f;

  for (int k0 = 0; k0 < SS; k0 += 32) {
    // stage alpha tile [128 x 32] fp32 -> f16
    for (int i = 0; i < 2; ++i) {
      int c = t + i * 256;                   // 512 chunks of 8 floats
      int row = c >> 2, off = (c & 3) << 3;
      float4 f0 = *(const float4*)&P[(size_t)(tq + row) * SS + k0 + off];
      float4 f1 = *(const float4*)&P[(size_t)(tq + row) * SS + k0 + off + 4];
      PackH u;
      u.h[0] = (_Float16)f0.x; u.h[1] = (_Float16)f0.y;
      u.h[2] = (_Float16)f0.z; u.h[3] = (_Float16)f0.w;
      u.h[4] = (_Float16)f1.x; u.h[5] = (_Float16)f1.y;
      u.h[6] = (_Float16)f1.z; u.h[7] = (_Float16)f1.w;
      *(float4*)&As[row * 32 + off] = u.f;
    }
    // stage V tile [64 d x 32 j]
    { int row = t >> 2, off = (t & 3) << 3;
      *(float4*)&Bs[row * 32 + off] = *(const float4*)&V[(size_t)row * SS + k0 + off]; }
    if (k0 + 32 < SS)
      __builtin_prefetch(&P[(size_t)(tq + (t >> 1)) * SS + k0 + 32], 0, 1);
    __syncthreads();

    v16h afrag, bfrag[4];
    {
      const _Float16* p = &As[(wq + l16) * 32];
      F16x8 u; u.f[0] = *(const float4*)(p + hi * 8);
               u.f[1] = *(const float4*)(p + 16 + hi * 8);
      afrag = u.v;
    }
    for (int j = 0; j < 4; ++j) {
      const _Float16* p = &Bs[(j * 16 + l16) * 32];
      F16x8 u; u.f[0] = *(const float4*)(p + hi * 8);
               u.f[1] = *(const float4*)(p + 16 + hi * 8);
      bfrag[j] = u.v;
    }
    for (int j = 0; j < 4; ++j)
      acc[j] = __builtin_amdgcn_wmma_f32_16x16x32_f16(
          false, afrag, false, bfrag[j], (short)0, acc[j], false, false);
    __syncthreads();
  }

  // context raw-reshape is identity: Ch flat = g*65536 + q*64 + d
  for (int j = 0; j < 4; ++j) {
    int col = j * 16 + l16;                 // d in [0,64)
    for (int r = 0; r < 8; ++r) {
      int qrow = tq + wq + r + hi * 8;      // q in [0,1024)
      Ch[(size_t)g * GELEM + (size_t)qrow * DHD + col] = (_Float16)acc[j][r];
    }
  }
}

// ---------------- out-proj + bias + residual (fp32, TDM double-buffered) ----------------
__global__ __launch_bounds__(256) void oproj_gemm(
    const _Float16* __restrict__ C,   // [8192,1024] f16
    const _Float16* __restrict__ Wo,  // [1024,1024] f16
    const float* __restrict__ bo, const float* __restrict__ Xres,
    float* __restrict__ Ptmp) {
  const int tm = blockIdx.x * 128;
  const int tn = blockIdx.y * 128;

  __shared__ alignas(16) _Float16 As[2][128 * 32];
  __shared__ alignas(16) _Float16 Bs[2][128 * 32];

  const int t = threadIdx.x;
  const int lane = t & 31, wave = t >> 5;
  const int wm = (wave >> 1) * 32;
  const int wn = (wave & 1) * 64;
  const int hi = (lane >= 16) ? 1 : 0;
  const int l16 = lane & 15;

  v8f acc[2][4];
  for (int i = 0; i < 2; ++i) for (int j = 0; j < 4; ++j) acc[i][j] = (v8f)0.0f;

  if (wave == 0) {
    tdm_load_tile_f16(lds_addr_of(As[0]), &C[(size_t)tm * DD],  DD, MTOT, DD, 32, 128);
    tdm_load_tile_f16(lds_addr_of(Bs[0]), &Wo[(size_t)tn * DD], DD, DD,   DD, 32, 128);
  }

  for (int it = 0; it < DD / 32; ++it) {
    const int k0 = it * 32;
    const int cur = it & 1;
    if (wave == 0) {
      if (k0 + 32 < DD) {
        tdm_load_tile_f16(lds_addr_of(As[cur ^ 1]), &C[(size_t)tm * DD + k0 + 32],  DD, MTOT, DD, 32, 128);
        tdm_load_tile_f16(lds_addr_of(Bs[cur ^ 1]), &Wo[(size_t)tn * DD + k0 + 32], DD, DD,   DD, 32, 128);
        __builtin_amdgcn_s_wait_tensorcnt(2);
      } else {
        __builtin_amdgcn_s_wait_tensorcnt(0);
      }
    }
    __syncthreads();

    v16h afrag[2], bfrag[4];
    for (int i = 0; i < 2; ++i) {
      const _Float16* p = &As[cur][(wm + i * 16 + l16) * 32];
      F16x8 u; u.f[0] = *(const float4*)(p + hi * 8);
               u.f[1] = *(const float4*)(p + 16 + hi * 8);
      afrag[i] = u.v;
    }
    for (int j = 0; j < 4; ++j) {
      const _Float16* p = &Bs[cur][(wn + j * 16 + l16) * 32];
      F16x8 u; u.f[0] = *(const float4*)(p + hi * 8);
               u.f[1] = *(const float4*)(p + 16 + hi * 8);
      bfrag[j] = u.v;
    }
    for (int i = 0; i < 2; ++i)
      for (int j = 0; j < 4; ++j)
        acc[i][j] = __builtin_amdgcn_wmma_f32_16x16x32_f16(
            false, afrag[i], false, bfrag[j], (short)0, acc[i][j], false, false);
    __syncthreads();
  }

  for (int i = 0; i < 2; ++i)
    for (int j = 0; j < 4; ++j) {
      int col = tn + wn + j * 16 + l16;
      float bcol = bo[col];
      for (int r = 0; r < 8; ++r) {
        int row = tm + wm + i * 16 + r + hi * 8;
        size_t idx = (size_t)row * DD + col;
        Ptmp[idx] = acc[i][j][r] + bcol + Xres[idx];
      }
    }
}

// ---------------- LayerNorm ----------------
__global__ __launch_bounds__(256) void ln_kernel(
    const float* __restrict__ P, const float* __restrict__ gamma,
    const float* __restrict__ beta, float* __restrict__ out) {
  const int row = blockIdx.x;
  const int t = threadIdx.x;
  const float* p = P + (size_t)row * DD;
  float4 v = *(const float4*)&p[t * 4];
  float s  = v.x + v.y + v.z + v.w;
  float s2 = v.x * v.x + v.y * v.y + v.z * v.z + v.w * v.w;
  for (int o = 16; o; o >>= 1) { s += __shfl_xor(s, o, 32); s2 += __shfl_xor(s2, o, 32); }
  __shared__ float w1[8], w2[8];
  if ((t & 31) == 0) { w1[t >> 5] = s; w2[t >> 5] = s2; }
  __syncthreads();
  float ts = 0.f, ts2 = 0.f;
  for (int i = 0; i < 8; ++i) { ts += w1[i]; ts2 += w2[i]; }
  float mu  = ts * (1.0f / DD);
  float var = ts2 * (1.0f / DD) - mu * mu;
  float inv = rsqrtf(var + LN_EPS);
  float4 g4 = *(const float4*)&gamma[t * 4];
  float4 b4 = *(const float4*)&beta[t * 4];
  float4 o;
  o.x = (v.x - mu) * inv * g4.x + b4.x;
  o.y = (v.y - mu) * inv * g4.y + b4.y;
  o.z = (v.z - mu) * inv * g4.z + b4.z;
  o.w = (v.w - mu) * inv * g4.w + b4.w;
  *(float4*)&out[(size_t)row * DD + t * 4] = o;
}

// ---------------- host launcher ----------------
extern "C" void kernel_launch(void* const* d_in, const int* in_sizes, int n_in,
                              void* d_out, int out_size, void* d_ws, size_t ws_size,
                              hipStream_t stream) {
  const float* x     = (const float*)d_in[0];
  const float* Wq    = (const float*)d_in[1];
  const float* bq    = (const float*)d_in[2];
  const float* Wk    = (const float*)d_in[3];
  const float* bk    = (const float*)d_in[4];
  const float* Wv    = (const float*)d_in[5];
  const float* bv    = (const float*)d_in[6];
  const float* Wo    = (const float*)d_in[7];
  const float* bo    = (const float*)d_in[8];
  const float* gamma = (const float*)d_in[9];
  const float* beta  = (const float*)d_in[10];

  float* out   = (float*)d_out;               // [8192*1024]
  float* alpha = out + (size_t)MTOT * DD;     // [128*1024*1024]

  // workspace layout (bytes)
  char* ws = (char*)d_ws;
  _Float16* xh  = (_Float16*)(ws);                       // 16 MiB
  _Float16* wh  = (_Float16*)(ws + (16u << 20));         // 3x2 MiB (q,k,v) + 2 MiB (o)
  _Float16* woh = wh + (size_t)3 * DD * DD;
  _Float16* Qh  = (_Float16*)(ws + (24u << 20));         // 16 MiB
  _Float16* Kh  = (_Float16*)(ws + (40u << 20));         // 16 MiB
  _Float16* Vt  = (_Float16*)(ws + (56u << 20));         // 16 MiB
  _Float16* Ch  = (_Float16*)(ws + (72u << 20));         // 16 MiB
  float*    Pt  = (float*)   (ws + (88u << 20));         // 32 MiB

  // 1) convert to f16
  cvt_f32_f16<<<4096, 256, 0, stream>>>(x,  xh,        MTOT * DD);
  cvt_f32_f16<<<1024, 256, 0, stream>>>(Wq, wh,              DD * DD);
  cvt_f32_f16<<<1024, 256, 0, stream>>>(Wk, wh + DD * DD,    DD * DD);
  cvt_f32_f16<<<1024, 256, 0, stream>>>(Wv, wh + 2 * DD * DD, DD * DD);
  cvt_f32_f16<<<1024, 256, 0, stream>>>(Wo, woh,             DD * DD);

  // 2) QKV projections (z selects weight / output; V stored transposed-in-group)
  qkv_gemm<<<dim3(MTOT / 128, DD / 128, 3), 256, 0, stream>>>(
      xh, wh, bq, bk, bv, Qh, Kh, Vt);

  // 3) scores into alpha region
  scores_kernel<<<dim3(SS / 128, SS / 128, GRP), 256, 0, stream>>>(Qh, Kh, alpha);

  // 4) softmax in place (1 wave per row, 8 rows per block)
  softmax_kernel<<<(GRP * SS) / 8, 256, 0, stream>>>(alpha);

  // 5) context = alpha @ V
  pv_kernel<<<dim3(SS / 128, GRP), 256, 0, stream>>>(alpha, Vt, Ch);

  // 6) out projection + residual, then LayerNorm
  oproj_gemm<<<dim3(MTOT / 128, DD / 128), 256, 0, stream>>>(Ch, woh, bo, x, Pt);
  ln_kernel<<<MTOT, 256, 0, stream>>>(Pt, gamma, beta, out);
}
